// SwitchMoELayer_61229053772309
// MI455X (gfx1250) — matrix operand order repaired and checked by reading, not compiled, not run on hardware
//
#include <hip/hip_runtime.h>
#include <math.h>

// Problem constants (from reference): B=4, S=2048 -> T=8192 tokens
constexpr int T = 8192;
constexpr int D = 1024;
constexpr int H = 4096;
constexpr int E = 8;

typedef __attribute__((ext_vector_type(16))) __bf16 v16bf;
typedef __attribute__((ext_vector_type(8)))  float  v8f;

union Frag { unsigned int u[8]; v16bf v; };

// ---------------- workspace layout (bytes) ----------------
// W1BF holds w1^T per expert: [E][H][D] bf16.  W2BF holds w2^T: [E][D][H] bf16.
constexpr size_t OFF_XBF   = 0;                                   // T*D bf16      = 16 MB
constexpr size_t OFF_W1BF  = OFF_XBF  + (size_t)T * D * 2;        // E*H*D bf16    = 64 MB
constexpr size_t OFF_W2BF  = OFF_W1BF + (size_t)E * D * H * 2;    // E*D*H bf16    = 64 MB
constexpr size_t OFF_HBF   = OFF_W2BF + (size_t)E * H * D * 2;    // T*H bf16      = 64 MB
constexpr size_t OFF_PROBS = OFF_HBF  + (size_t)T * H * 2;        // T*E f32       = 256 KB
constexpr size_t OFF_GATE  = OFF_PROBS + (size_t)T * E * 4;       // T f32
constexpr size_t OFF_SEL   = OFF_GATE  + (size_t)T * 4;           // T i32
constexpr size_t OFF_TOK   = OFF_SEL   + (size_t)T * 4;           // T i32
constexpr size_t OFF_CNT   = OFF_TOK   + (size_t)T * 4;           // E i32
constexpr size_t OFF_CUR   = OFF_CNT   + 256;                     // E i32
constexpr size_t OFF_OFFS  = OFF_CUR   + 256;                     // E i32

// ---------------- small utility kernels ----------------
__global__ __launch_bounds__(32) void init_kernel(int* counts, int* cursors) {
    int i = threadIdx.x;
    if (i < E) { counts[i] = 0; cursors[i] = 0; }
}

__global__ __launch_bounds__(256) void tobf16_kernel(const float* __restrict__ src,
                                                     __bf16* __restrict__ dst, int n) {
    int i = (blockIdx.x * 256 + threadIdx.x) * 4;
    if (i + 3 < n) {
        float4 v = *(const float4*)(src + i);
        dst[i + 0] = (__bf16)v.x;
        dst[i + 1] = (__bf16)v.y;
        dst[i + 2] = (__bf16)v.z;
        dst[i + 3] = (__bf16)v.w;
    }
}

// Per-expert transpose + fp32->bf16: src [K,N] fp32 -> dst [N,K] bf16.
__global__ __launch_bounds__(256) void transpose_bf16_kernel(const float* __restrict__ src,
                                                             __bf16* __restrict__ dst,
                                                             int K, int N) {
    __shared__ float tile[32][33];
    const float* S = src + (size_t)blockIdx.z * K * N;
    __bf16* Dt = dst + (size_t)blockIdx.z * K * N;
    int k0 = blockIdx.y * 32, n0 = blockIdx.x * 32;
    int tx = threadIdx.x & 31, ty = threadIdx.x >> 5;  // 32 x 8
    #pragma unroll
    for (int r = ty; r < 32; r += 8)
        tile[r][tx] = S[(size_t)(k0 + r) * N + n0 + tx];
    __syncthreads();
    #pragma unroll
    for (int r = ty; r < 32; r += 8)
        Dt[(size_t)(n0 + r) * K + k0 + tx] = (__bf16)tile[tx][r];
}

// One wave per token: coalesced reads of x, shuffle reduction, softmax + top-1.
__global__ __launch_bounds__(256) void router_kernel(const float* __restrict__ x,
                                                     const float* __restrict__ rw,
                                                     const float* __restrict__ rb,
                                                     float* __restrict__ probs,
                                                     float* __restrict__ gate,
                                                     int* __restrict__ sel,
                                                     int* __restrict__ counts) {
    int wave = threadIdx.x >> 5;
    int lane = threadIdx.x & 31;
    int t = blockIdx.x * 8 + wave;
    if (t >= T) return;
    const float* xr = x + (size_t)t * D;
    float acc[E] = {};
    for (int d = lane; d < D; d += 32) {
        float xv = xr[d];
        #pragma unroll
        for (int e = 0; e < E; ++e) acc[e] += xv * rw[d * E + e];
    }
    #pragma unroll
    for (int off = 16; off > 0; off >>= 1) {
        #pragma unroll
        for (int e = 0; e < E; ++e) acc[e] += __shfl_xor(acc[e], off, 32);
    }
    if (lane == 0) {
        float lg[E];
        #pragma unroll
        for (int e = 0; e < E; ++e) lg[e] = acc[e] + rb[e];
        int best = 0;
        float m = lg[0];
        #pragma unroll
        for (int e = 1; e < E; ++e) { if (lg[e] > m) { m = lg[e]; best = e; } }
        float s = 0.f, p[E];
        #pragma unroll
        for (int e = 0; e < E; ++e) { p[e] = __expf(lg[e] - m); s += p[e]; }
        float inv = 1.0f / s;
        #pragma unroll
        for (int e = 0; e < E; ++e) probs[(size_t)t * E + e] = p[e] * inv;
        gate[t] = p[best] * inv;
        sel[t] = best;
        atomicAdd(&counts[best], 1);
    }
}

__global__ __launch_bounds__(32) void prefix_kernel(const int* counts, int* offsets) {
    if (threadIdx.x == 0) {
        int s = 0;
        for (int e = 0; e < E; ++e) { offsets[e] = s; s += counts[e]; }
    }
}

__global__ __launch_bounds__(256) void scatter_kernel(const int* __restrict__ sel,
                                                      const int* __restrict__ offsets,
                                                      int* __restrict__ cursors,
                                                      int* __restrict__ tokens) {
    int t = blockIdx.x * 256 + threadIdx.x;
    if (t >= T) return;
    int e = sel[t];
    int slot = atomicAdd(&cursors[e], 1);
    tokens[offsets[e] + slot] = t;
}

// ---------------- WMMA GEMM tiles ----------------
// Block tile 128(M) x 128(N), K-chunk 32.  8 waves: 4 along M x 2 along N,
// each wave owns 32x64 = 2x4 WMMA 16x16 f32 accumulators.
// Both A (tokens x K) and B (w^T: ncols x K) are staged identically:
// register-pipelined global_load_b128 -> ds_store_b128, LDS layout [row][k].

__device__ __forceinline__ float gelu_tanh(float v) {
    float v3 = v * v * v;
    return 0.5f * v * (1.0f + tanhf(0.79788456080286535588f * (v + 0.044715f * v3)));
}

// GEMM1: h = gelu(x[tok] @ w1[e] + b1[e])  (K = D = 1024, N = H = 4096)
__global__ __launch_bounds__(256) void gemm1_kernel(const __bf16* __restrict__ xbf,
                                                    const __bf16* __restrict__ w1t,
                                                    const float* __restrict__ b1,
                                                    const int* __restrict__ tokens,
                                                    const int* __restrict__ counts,
                                                    const int* __restrict__ offsets,
                                                    __bf16* __restrict__ hbf) {
    const int e = blockIdx.z;
    const int cnt = counts[e];
    const int m0 = blockIdx.y * 128;
    if (m0 >= cnt) return;
    const int n0 = blockIdx.x * 128;
    const int base = offsets[e];

    __shared__ __attribute__((aligned(16))) __bf16 As[128 * 32];
    __shared__ __attribute__((aligned(16))) __bf16 Bs[128 * 32];  // [n][k]
    __shared__ int toks[128];

    const int tid = threadIdx.x;
    if (tid < 128) {
        int m = m0 + tid;
        toks[tid] = tokens[base + (m < cnt ? m : 0)];
    }
    __syncthreads();

    const int lane = tid & 31;
    const int wave = tid >> 5;
    const int waveM = wave & 3;
    const int waveN = wave >> 2;
    const int hf = lane >> 4;
    const int l16 = lane & 15;

    // Staging geometry: chunk = tid + p*256; row = chunk/4; c4 = chunk%4 (16B each).
    const int rowA0 = tid >> 2, rowA1 = (tid + 256) >> 2;
    const int c4a = (tid & 3) * 8;
    const __bf16* Wt = w1t + (size_t)e * H * D;  // [H][D]
    const __bf16* aSrc0 = xbf + (size_t)toks[rowA0] * D + c4a;
    const __bf16* aSrc1 = xbf + (size_t)toks[rowA1] * D + c4a;
    const __bf16* bSrc0 = Wt + (size_t)(n0 + rowA0) * D + c4a;
    const __bf16* bSrc1 = Wt + (size_t)(n0 + rowA1) * D + c4a;

    v8f acc[2][4] = {};
    uint4 aReg0 = *(const uint4*)aSrc0, aReg1 = *(const uint4*)aSrc1;
    uint4 bReg0 = *(const uint4*)bSrc0, bReg1 = *(const uint4*)bSrc1;

    for (int k0 = 0; k0 < D; k0 += 32) {
        *(uint4*)(&As[rowA0 * 32 + c4a]) = aReg0;
        *(uint4*)(&As[rowA1 * 32 + c4a]) = aReg1;
        *(uint4*)(&Bs[rowA0 * 32 + c4a]) = bReg0;
        *(uint4*)(&Bs[rowA1 * 32 + c4a]) = bReg1;
        __syncthreads();
        if (k0 + 32 < D) {  // issue next tile's loads; latency hides under WMMAs
            aReg0 = *(const uint4*)(aSrc0 + k0 + 32);
            aReg1 = *(const uint4*)(aSrc1 + k0 + 32);
            bReg0 = *(const uint4*)(bSrc0 + k0 + 32);
            bReg1 = *(const uint4*)(bSrc1 + k0 + 32);
            __builtin_prefetch(bSrc0 + k0 + 64, 0, 1);
        }
        Frag afr[2], bfr[4];
        #pragma unroll
        for (int i = 0; i < 2; ++i) {
            int mrow = waveM * 32 + i * 16 + l16;
            #pragma unroll
            for (int v = 0; v < 8; ++v) {
                int kk = (v >> 2) * 16 + hf * 8 + (v & 3) * 2;
                afr[i].u[v] = *(const unsigned int*)(&As[mrow * 32 + kk]);
            }
        }
        #pragma unroll
        for (int j = 0; j < 4; ++j) {
            int ncol = waveN * 64 + j * 16 + l16;
            #pragma unroll
            for (int v = 0; v < 8; ++v) {
                int kk = (v >> 2) * 16 + hf * 8 + (v & 3) * 2;
                bfr[j].u[v] = *(const unsigned int*)(&Bs[ncol * 32 + kk]);
            }
        }
        #pragma unroll
        for (int i = 0; i < 2; ++i)
            #pragma unroll
            for (int j = 0; j < 4; ++j)
                acc[i][j] = __builtin_amdgcn_wmma_f32_16x16x32_bf16(
                    false, afr[i].v, false, bfr[j].v, (short)0, acc[i][j], false, false);
        __syncthreads();
    }

    const float* b1e = b1 + (size_t)e * H;
    #pragma unroll
    for (int j = 0; j < 4; ++j) {
        int ncol = n0 + waveN * 64 + j * 16 + l16;
        float bias = b1e[ncol];
        #pragma unroll
        for (int i = 0; i < 2; ++i) {
            #pragma unroll
            for (int r = 0; r < 8; ++r) {
                int mloc = waveM * 32 + i * 16 + hf * 8 + r;
                if (m0 + mloc < cnt) {
                    float g = gelu_tanh(acc[i][j][r] + bias);
                    hbf[(size_t)(base + m0 + mloc) * H + ncol] = (__bf16)g;
                }
            }
        }
    }
}

// GEMM2: out[tok] = gate[tok] * (h @ w2[e] + b2[e])  (K = H = 4096, N = D = 1024)
__global__ __launch_bounds__(256) void gemm2_kernel(const __bf16* __restrict__ hbf,
                                                    const __bf16* __restrict__ w2t,
                                                    const float* __restrict__ b2,
                                                    const int* __restrict__ tokens,
                                                    const int* __restrict__ counts,
                                                    const int* __restrict__ offsets,
                                                    const float* __restrict__ gate,
                                                    float* __restrict__ out) {
    const int e = blockIdx.z;
    const int cnt = counts[e];
    const int m0 = blockIdx.y * 128;
    if (m0 >= cnt) return;
    const int n0 = blockIdx.x * 128;
    const int base = offsets[e];

    __shared__ __attribute__((aligned(16))) __bf16 As[128 * 32];
    __shared__ __attribute__((aligned(16))) __bf16 Bs[128 * 32];  // [n][k]
    __shared__ int toks[128];
    __shared__ float gts[128];

    const int tid = threadIdx.x;
    if (tid < 128) {
        int m = m0 + tid;
        int t = tokens[base + (m < cnt ? m : 0)];
        toks[tid] = t;
        gts[tid] = gate[t];
    }
    __syncthreads();

    const int lane = tid & 31;
    const int wave = tid >> 5;
    const int waveM = wave & 3;
    const int waveN = wave >> 2;
    const int hf = lane >> 4;
    const int l16 = lane & 15;

    const int rowA0 = tid >> 2, rowA1 = (tid + 256) >> 2;
    const int c4a = (tid & 3) * 8;
    const int rA0 = (m0 + rowA0 < cnt) ? rowA0 : 0;  // clamp rows past count
    const int rA1 = (m0 + rowA1 < cnt) ? rowA1 : 0;
    const __bf16* Wt = w2t + (size_t)e * D * H;  // [D][H]
    const __bf16* aSrc0 = hbf + (size_t)(base + m0 + rA0) * H + c4a;
    const __bf16* aSrc1 = hbf + (size_t)(base + m0 + rA1) * H + c4a;
    const __bf16* bSrc0 = Wt + (size_t)(n0 + rowA0) * H + c4a;
    const __bf16* bSrc1 = Wt + (size_t)(n0 + rowA1) * H + c4a;

    v8f acc[2][4] = {};
    uint4 aReg0 = *(const uint4*)aSrc0, aReg1 = *(const uint4*)aSrc1;
    uint4 bReg0 = *(const uint4*)bSrc0, bReg1 = *(const uint4*)bSrc1;

    for (int k0 = 0; k0 < H; k0 += 32) {
        *(uint4*)(&As[rowA0 * 32 + c4a]) = aReg0;
        *(uint4*)(&As[rowA1 * 32 + c4a]) = aReg1;
        *(uint4*)(&Bs[rowA0 * 32 + c4a]) = bReg0;
        *(uint4*)(&Bs[rowA1 * 32 + c4a]) = bReg1;
        __syncthreads();
        if (k0 + 32 < H) {
            aReg0 = *(const uint4*)(aSrc0 + k0 + 32);
            aReg1 = *(const uint4*)(aSrc1 + k0 + 32);
            bReg0 = *(const uint4*)(bSrc0 + k0 + 32);
            bReg1 = *(const uint4*)(bSrc1 + k0 + 32);
            __builtin_prefetch(bSrc0 + k0 + 64, 0, 1);
        }
        Frag afr[2], bfr[4];
        #pragma unroll
        for (int i = 0; i < 2; ++i) {
            int mrow = waveM * 32 + i * 16 + l16;
            #pragma unroll
            for (int v = 0; v < 8; ++v) {
                int kk = (v >> 2) * 16 + hf * 8 + (v & 3) * 2;
                afr[i].u[v] = *(const unsigned int*)(&As[mrow * 32 + kk]);
            }
        }
        #pragma unroll
        for (int j = 0; j < 4; ++j) {
            int ncol = waveN * 64 + j * 16 + l16;
            #pragma unroll
            for (int v = 0; v < 8; ++v) {
                int kk = (v >> 2) * 16 + hf * 8 + (v & 3) * 2;
                bfr[j].u[v] = *(const unsigned int*)(&Bs[ncol * 32 + kk]);
            }
        }
        #pragma unroll
        for (int i = 0; i < 2; ++i)
            #pragma unroll
            for (int j = 0; j < 4; ++j)
                acc[i][j] = __builtin_amdgcn_wmma_f32_16x16x32_bf16(
                    false, afr[i].v, false, bfr[j].v, (short)0, acc[i][j], false, false);
        __syncthreads();
    }

    const float* b2e = b2 + (size_t)e * D;
    #pragma unroll
    for (int j = 0; j < 4; ++j) {
        int ncol = n0 + waveN * 64 + j * 16 + l16;
        float bias = b2e[ncol];
        #pragma unroll
        for (int i = 0; i < 2; ++i) {
            #pragma unroll
            for (int r = 0; r < 8; ++r) {
                int mloc = waveM * 32 + i * 16 + hf * 8 + r;
                if (m0 + mloc < cnt) {
                    int t = toks[mloc];
                    out[(size_t)t * D + ncol] = gts[mloc] * (acc[i][j][r] + bias);
                }
            }
        }
    }
}

// Deterministic aux-loss reduction (fixed tree over probs buffer).
__global__ __launch_bounds__(256) void aux_kernel(const float* __restrict__ probs,
                                                  const int* __restrict__ counts,
                                                  float* __restrict__ loss_out) {
    __shared__ float red[256];
    float s[E] = {};
    for (int t = threadIdx.x; t < T; t += 256) {
        #pragma unroll
        for (int e = 0; e < E; ++e) s[e] += probs[(size_t)t * E + e];
    }
    float total = 0.f;
    for (int e = 0; e < E; ++e) {
        red[threadIdx.x] = s[e];
        __syncthreads();
        for (int off = 128; off > 0; off >>= 1) {
            if (threadIdx.x < off) red[threadIdx.x] += red[threadIdx.x + off];
            __syncthreads();
        }
        if (threadIdx.x == 0)
            total += (red[0] / (float)T) * ((float)counts[e] / (float)T);
        __syncthreads();
    }
    if (threadIdx.x == 0) *loss_out = 0.01f * (float)E * total;
}

// ---------------- launcher ----------------
extern "C" void kernel_launch(void* const* d_in, const int* in_sizes, int n_in,
                              void* d_out, int out_size, void* d_ws, size_t ws_size,
                              hipStream_t stream) {
    const float* x  = (const float*)d_in[0];
    const float* rw = (const float*)d_in[1];
    const float* rb = (const float*)d_in[2];
    const float* w1 = (const float*)d_in[3];
    const float* b1 = (const float*)d_in[4];
    const float* w2 = (const float*)d_in[5];
    const float* b2 = (const float*)d_in[6];
    float* out = (float*)d_out;

    char* ws = (char*)d_ws;
    __bf16* xbf   = (__bf16*)(ws + OFF_XBF);
    __bf16* w1t   = (__bf16*)(ws + OFF_W1BF);
    __bf16* w2t   = (__bf16*)(ws + OFF_W2BF);
    __bf16* hbf   = (__bf16*)(ws + OFF_HBF);
    float*  probs = (float*)(ws + OFF_PROBS);
    float*  gate  = (float*)(ws + OFF_GATE);
    int*    sel   = (int*)(ws + OFF_SEL);
    int*    toks  = (int*)(ws + OFF_TOK);
    int*    cnts  = (int*)(ws + OFF_CNT);
    int*    curs  = (int*)(ws + OFF_CUR);
    int*    offs  = (int*)(ws + OFF_OFFS);

    init_kernel<<<1, 32, 0, stream>>>(cnts, curs);
    tobf16_kernel<<<(T * D) / 1024, 256, 0, stream>>>(x, xbf, T * D);
    // w1 [E][D][H] -> w1t [E][H][D];  w2 [E][H][D] -> w2t [E][D][H]
    transpose_bf16_kernel<<<dim3(H / 32, D / 32, E), 256, 0, stream>>>(w1, w1t, D, H);
    transpose_bf16_kernel<<<dim3(D / 32, H / 32, E), 256, 0, stream>>>(w2, w2t, H, D);
    router_kernel<<<T / 8, 256, 0, stream>>>(x, rw, rb, probs, gate, sel, cnts);
    prefix_kernel<<<1, 32, 0, stream>>>(cnts, offs);
    scatter_kernel<<<T / 256, 256, 0, stream>>>(sel, offs, curs, toks);
    gemm1_kernel<<<dim3(H / 128, T / 128, E), 256, 0, stream>>>(xbf, w1t, b1, toks, cnts, offs, hbf);
    gemm2_kernel<<<dim3(D / 128, T / 128, E), 256, 0, stream>>>(hbf, w2t, b2, toks, cnts, offs, gate, out);
    aux_kernel<<<1, 256, 0, stream>>>(probs, cnts, out + (size_t)T * D);
}
